// DynamicFusionVFE_4999341933061
// MI455X (gfx1250) — matrix operand order repaired and checked by reading, not compiled, not run on hardware
//
#include <hip/hip_runtime.h>
#include <hip/hip_bf16.h>
#include <math.h>

// ---------------------------------------------------------------------------
// DynamicFusionVFE on gfx1250 (wave32, WMMA f32<-f16 16x16x32)
// ---------------------------------------------------------------------------

typedef __attribute__((ext_vector_type(16))) _Float16 v16h;
typedef __attribute__((ext_vector_type(8)))  _Float16 v8h;
typedef __attribute__((ext_vector_type(8)))  float    v8f;

#define VFE_VX 0.2f
#define VFE_VY 0.2f
#define VFE_VZ 4.0f
#define VFE_XO 0.1f      /* VX/2 + 0.0   */
#define VFE_YO (-39.9f)  /* VY/2 - 40.0  */
#define VFE_ZO (-1.0f)   /* VZ/2 - 3.0   */
#define BN_EPS 1e-3f

// ---------------------------------------------------------------- utilities
__global__ void vfe_zero_f32(float* p, long n) {
  long i = (long)blockIdx.x * blockDim.x + threadIdx.x;
  if (i < n) p[i] = 0.0f;
}
__global__ void vfe_zero_f16(_Float16* p, long n) {
  long i = (long)blockIdx.x * blockDim.x + threadIdx.x;
  if (i < n) p[i] = (_Float16)0.0f;
}

// --------------------------------------------------- pack W0 (11x64 -> B-layout)
// B-layout for one 32x16 slab: element e of lane l holds B[(l>>4)*16+e][coltile*16+(l&15)]
// W0s index: ((coltile)*32 + lane)*16 + e   ; K padded 11 -> 32 with zeros.
__global__ void vfe_pack_w0(const float* W0, _Float16* W0s) {
  int i = blockIdx.x * blockDim.x + threadIdx.x;         // 4*32*16 = 2048
  if (i >= 2048) return;
  int e = i & 15, lane = (i >> 4) & 31, t = i >> 9;
  int k = ((lane >> 4) << 4) + e;
  int n = (t << 4) + (lane & 15);
  float v = (k < 11) ? W0[k * 64 + n] : 0.0f;
  W0s[i] = (_Float16)v;
}

// --------------------------------------------------- pack W1 (128x128 -> B-layout)
// W1s index: ((coltile*4 + k0)*32 + lane)*16 + e ; k = k0*32 + (lane>>4)*16 + e
__global__ void vfe_pack_w1(const float* W1, _Float16* W1s) {
  int i = blockIdx.x * blockDim.x + threadIdx.x;         // 8*4*32*16 = 16384
  if (i >= 16384) return;
  int e = i & 15, lane = (i >> 4) & 31, k0 = (i >> 9) & 3, t = i >> 11;
  int k = k0 * 32 + ((lane >> 4) << 4) + e;
  int n = (t << 4) + (lane & 15);
  W1s[i] = (_Float16)W1[k * 128 + n];
}

// --------------------------------------------------- scatter mean accumulation
__global__ void vfe_scatter_sum(const float* feats, const int* seg,
                                float* vsum, float* vcnt, int N) {
  int n = blockIdx.x * blockDim.x + threadIdx.x;
  if (n >= N) return;
  int s = seg[n];
  atomicAdd(&vsum[s * 4 + 0], feats[n * 4 + 0]);
  atomicAdd(&vsum[s * 4 + 1], feats[n * 4 + 1]);
  atomicAdd(&vsum[s * 4 + 2], feats[n * 4 + 2]);
  atomicAdd(&vcnt[s], 1.0f);
}

// --------------------------------------------------- build x [Np,32] f16 (K-pad)
__global__ void vfe_build_x(const float* feats, const int* coors, const int* seg,
                            const float* vsum, const float* vcnt,
                            _Float16* x16, int N, int Np) {
  int n = blockIdx.x * blockDim.x + threadIdx.x;
  if (n >= Np) return;
  _Float16* row = x16 + (size_t)n * 32;
  if (n >= N) {
#pragma unroll
    for (int i = 0; i < 32; ++i) row[i] = (_Float16)0.0f;
    return;
  }
  float x  = feats[n * 4 + 0];
  float y  = feats[n * 4 + 1];
  float z  = feats[n * 4 + 2];
  float it = feats[n * 4 + 3];
  int s = seg[n];
  float inv = 1.0f / vcnt[s];
  float mx = vsum[s * 4 + 0] * inv;
  float my = vsum[s * 4 + 1] * inv;
  float mz = vsum[s * 4 + 2] * inv;
  float cxf = (float)coors[n * 4 + 3];
  float cyf = (float)coors[n * 4 + 2];
  float czf = (float)coors[n * 4 + 1];
  float v[11];
  v[0] = x; v[1] = y; v[2] = z; v[3] = it;
  v[4] = x - mx; v[5] = y - my; v[6] = z - mz;
  v[7] = x - (cxf * VFE_VX + VFE_XO);
  v[8] = y - (cyf * VFE_VY + VFE_YO);
  v[9] = z - (czf * VFE_VZ + VFE_ZO);
  v[10] = sqrtf(x * x + y * y + z * z);
#pragma unroll
  for (int i = 0; i < 11; ++i) row[i] = (_Float16)v[i];
#pragma unroll
  for (int i = 11; i < 32; ++i) row[i] = (_Float16)0.0f;
}

// --------------------------------------------------- WMMA helpers
__device__ inline v16h vfe_load_a(const _Float16* rowbase, int hi) {
  // A 16x32 layout: lane's row, contiguous chunk [hi*16 .. hi*16+15]
  v8h lo = *(const v8h*)(rowbase + hi * 16);
  v8h hh = *(const v8h*)(rowbase + hi * 16 + 8);
  v16h a;
#pragma unroll
  for (int i = 0; i < 8; ++i) { a[i] = lo[i]; a[i + 8] = hh[i]; }
  return a;
}

// =================================================== GEMM1 stats: x16 @ W0s
// grid.x over rowtile groups of 4 waves, grid.y = 4 col tiles, block = 128
__global__ void vfe_gemm1_stats(const _Float16* __restrict__ x16,
                                const _Float16* __restrict__ W0s,
                                float* __restrict__ gsum, float* __restrict__ gsq,
                                int rowTiles) {
  __shared__ float ssum[16], ssq[16];
  int lane = threadIdx.x & 31;
  int wave = threadIdx.x >> 5;
  int rowtile = blockIdx.x * 4 + wave;
  int t = blockIdx.y;                    // col tile (0..3)
  if (threadIdx.x < 16) { ssum[threadIdx.x] = 0.0f; ssq[threadIdx.x] = 0.0f; }
  __syncthreads();

  if (rowtile < rowTiles) {
    int hi = lane >> 4;
    const _Float16* arow = x16 + ((size_t)(rowtile * 16 + (lane & 15))) * 32;
    v16h a = vfe_load_a(arow, hi);
    v16h b = *(const v16h*)(W0s + ((size_t)(t * 32 + lane)) * 16);
    v8f acc = {};
    acc = __builtin_amdgcn_wmma_f32_16x16x32_f16(false, a, false, b,
                                                 (short)0, acc, false, false);
    float s = 0.0f, s2 = 0.0f;
#pragma unroll
    for (int r = 0; r < 8; ++r) { s += acc[r]; s2 += acc[r] * acc[r]; }
    s  += __shfl_xor(s, 16, 32);
    s2 += __shfl_xor(s2, 16, 32);
    if (lane < 16) { atomicAdd(&ssum[lane], s); atomicAdd(&ssq[lane], s2); }
  }
  __syncthreads();
  if (threadIdx.x < 16) {
    atomicAdd(&gsum[t * 16 + threadIdx.x], ssum[threadIdx.x]);
    atomicAdd(&gsq [t * 16 + threadIdx.x], ssq [threadIdx.x]);
  }
}

// =================================================== BN params: a=g*rsqrt(var+eps), c=b-mean*a
__global__ void vfe_bnparams(const float* __restrict__ stats,  // [sum C][sq C]
                             const float* __restrict__ g, const float* __restrict__ bb,
                             float* __restrict__ a, float* __restrict__ c,
                             int C, float n) {
  int i = blockIdx.x * blockDim.x + threadIdx.x;
  if (i >= C) return;
  float mean = stats[i] / n;
  float var  = stats[C + i] / n - mean * mean;
  float ai = g[i] * rsqrtf(var + BN_EPS);
  a[i] = ai;
  c[i] = bb[i] - mean * ai;
}

// =================================================== GEMM1 apply: BN+ReLU -> x1[:,0:64] f16 + atomicMax v0
__global__ void vfe_gemm1_apply(const _Float16* __restrict__ x16,
                                const _Float16* __restrict__ W0s,
                                const float* __restrict__ bna, const float* __restrict__ bnc,
                                const int* __restrict__ seg,
                                _Float16* __restrict__ x1, float* __restrict__ v0,
                                int N, int rowTiles) {
  int lane = threadIdx.x & 31;
  int wave = threadIdx.x >> 5;
  int rowtile = blockIdx.x * 4 + wave;
  int t = blockIdx.y;
  if (rowtile >= rowTiles) return;
  int hi = lane >> 4;
  const _Float16* arow = x16 + ((size_t)(rowtile * 16 + (lane & 15))) * 32;
  v16h a = vfe_load_a(arow, hi);
  v16h b = *(const v16h*)(W0s + ((size_t)(t * 32 + lane)) * 16);
  v8f acc = {};
  acc = __builtin_amdgcn_wmma_f32_16x16x32_f16(false, a, false, b,
                                               (short)0, acc, false, false);
  int col = t * 16 + (lane & 15);
  float sa = bna[col], sc = bnc[col];
#pragma unroll
  for (int r = 0; r < 8; ++r) {
    int row = rowtile * 16 + hi * 8 + r;
    if (row < N) {
      float val = fmaxf(sa * acc[r] + sc, 0.0f);
      x1[(size_t)row * 128 + col] = (_Float16)val;
      atomicMax((int*)&v0[(size_t)seg[row] * 64 + col], __float_as_int(val));
    }
  }
}

// --------------------------------------------------- gather concat: x1[:,64:128] = v0[seg]
__global__ void vfe_gather_v0(const float* __restrict__ v0, const int* __restrict__ seg,
                              _Float16* __restrict__ x1, int N) {
  long i = (long)blockIdx.x * blockDim.x + threadIdx.x;
  if (i >= (long)N * 64) return;
  int n = (int)(i >> 6);
  int j = (int)(i & 63);
  x1[(size_t)n * 128 + 64 + j] = (_Float16)v0[(size_t)seg[n] * 64 + j];
}

// =================================================== GEMM2 stats: x1 @ W1s (K=128, 4 steps)
__global__ void vfe_gemm2_stats(const _Float16* __restrict__ x1,
                                const _Float16* __restrict__ W1s,
                                float* __restrict__ gsum, float* __restrict__ gsq,
                                int rowTiles) {
  __shared__ float ssum[16], ssq[16];
  int lane = threadIdx.x & 31;
  int wave = threadIdx.x >> 5;
  int rowtile = blockIdx.x * 4 + wave;
  int t = blockIdx.y;                    // col tile (0..7)
  if (threadIdx.x < 16) { ssum[threadIdx.x] = 0.0f; ssq[threadIdx.x] = 0.0f; }
  __syncthreads();

  if (rowtile < rowTiles) {
    int hi = lane >> 4;
    const _Float16* arow = x1 + ((size_t)(rowtile * 16 + (lane & 15))) * 128;
    v8f acc = {};
#pragma unroll
    for (int k0 = 0; k0 < 4; ++k0) {
      v16h a = vfe_load_a(arow + k0 * 32, hi);
      v16h b = *(const v16h*)(W1s + ((size_t)((t * 4 + k0) * 32 + lane)) * 16);
      acc = __builtin_amdgcn_wmma_f32_16x16x32_f16(false, a, false, b,
                                                   (short)0, acc, false, false);
    }
    float s = 0.0f, s2 = 0.0f;
#pragma unroll
    for (int r = 0; r < 8; ++r) { s += acc[r]; s2 += acc[r] * acc[r]; }
    s  += __shfl_xor(s, 16, 32);
    s2 += __shfl_xor(s2, 16, 32);
    if (lane < 16) { atomicAdd(&ssum[lane], s); atomicAdd(&ssq[lane], s2); }
  }
  __syncthreads();
  if (threadIdx.x < 16) {
    atomicAdd(&gsum[t * 16 + threadIdx.x], ssum[threadIdx.x]);
    atomicAdd(&gsq [t * 16 + threadIdx.x], ssq [threadIdx.x]);
  }
}

// =================================================== GEMM2 apply: BN+ReLU -> atomicMax out[M,128]
__global__ void vfe_gemm2_apply(const _Float16* __restrict__ x1,
                                const _Float16* __restrict__ W1s,
                                const float* __restrict__ bna, const float* __restrict__ bnc,
                                const int* __restrict__ seg,
                                float* __restrict__ out, int N, int rowTiles) {
  int lane = threadIdx.x & 31;
  int wave = threadIdx.x >> 5;
  int rowtile = blockIdx.x * 4 + wave;
  int t = blockIdx.y;
  if (rowtile >= rowTiles) return;
  int hi = lane >> 4;
  const _Float16* arow = x1 + ((size_t)(rowtile * 16 + (lane & 15))) * 128;
  v8f acc = {};
#pragma unroll
  for (int k0 = 0; k0 < 4; ++k0) {
    v16h a = vfe_load_a(arow + k0 * 32, hi);
    v16h b = *(const v16h*)(W1s + ((size_t)((t * 4 + k0) * 32 + lane)) * 16);
    acc = __builtin_amdgcn_wmma_f32_16x16x32_f16(false, a, false, b,
                                                 (short)0, acc, false, false);
  }
  int col = t * 16 + (lane & 15);
  float sa = bna[col], sc = bnc[col];
#pragma unroll
  for (int r = 0; r < 8; ++r) {
    int row = rowtile * 16 + hi * 8 + r;
    if (row < N) {
      float val = fmaxf(sa * acc[r] + sc, 0.0f);
      atomicMax((int*)&out[(size_t)seg[row] * 128 + col], __float_as_int(val));
    }
  }
}

// --------------------------------------------------- passthrough voxel coords
__global__ void vfe_copy_coors(const int* vc, int* dst, int n) {
  int i = blockIdx.x * blockDim.x + threadIdx.x;
  if (i < n) dst[i] = vc[i];
}

// ---------------------------------------------------------------------------
extern "C" void kernel_launch(void* const* d_in, const int* in_sizes, int n_in,
                              void* d_out, int out_size, void* d_ws, size_t ws_size,
                              hipStream_t stream) {
  const float* feats = (const float*)d_in[0];
  const int*   coors = (const int*)  d_in[1];
  const int*   seg   = (const int*)  d_in[2];
  const int*   vcrs  = (const int*)  d_in[3];
  const float* W0    = (const float*)d_in[4];
  const float* g0    = (const float*)d_in[5];
  const float* b0    = (const float*)d_in[6];
  const float* W1    = (const float*)d_in[7];
  const float* g1    = (const float*)d_in[8];
  const float* b1    = (const float*)d_in[9];

  const int N  = in_sizes[2];
  const int M  = in_sizes[3] / 4;
  const int Np = (N + 15) & ~15;
  const int rowTiles = Np / 16;

  // ---- workspace layout
  char* ws = (char*)d_ws;
  size_t off = 0;
  auto take = [&](size_t bytes) -> char* {
    char* p = ws + off;
    off = (off + bytes + 255) & ~(size_t)255;
    return p;
  };
  float*     vsum  = (float*)    take((size_t)M * 4 * sizeof(float));
  float*     vcnt  = (float*)    take((size_t)M * sizeof(float));
  float*     stats = (float*)    take(768 * sizeof(float));
  float*     v0    = (float*)    take((size_t)M * 64 * sizeof(float));
  _Float16*  x16   = (_Float16*) take((size_t)Np * 32 * sizeof(_Float16));
  _Float16*  x1    = (_Float16*) take((size_t)Np * 128 * sizeof(_Float16));
  _Float16*  W0s   = (_Float16*) take(2048 * sizeof(_Float16));
  _Float16*  W1s   = (_Float16*) take(16384 * sizeof(_Float16));
  (void)ws_size;

  // stats region: [sum0 64][sq0 64][a0 64][c0 64][sum1 128][sq1 128][a1 128][c1 128]
  float* stats0 = stats;           // sum0+sq0 (contiguous 128)
  float* a0p    = stats + 128;
  float* c0p    = stats + 192;
  float* stats1 = stats + 256;     // sum1+sq1 (contiguous 256)
  float* a1p    = stats + 512;
  float* c1p    = stats + 640;

  float* outF = (float*)d_out;                 // [M,128]
  int*   outC = (int*)d_out + (size_t)M * 128; // [M,4]

  const int B = 256;
  auto gr = [&](long n) { return dim3((unsigned)((n + B - 1) / B)); };

  // 1. zero accumulators / outputs
  vfe_zero_f32<<<gr((long)M * 4), B, 0, stream>>>(vsum, (long)M * 4);
  vfe_zero_f32<<<gr(M), B, 0, stream>>>(vcnt, M);
  vfe_zero_f32<<<gr(768), B, 0, stream>>>(stats, 768);
  vfe_zero_f32<<<gr((long)M * 64), B, 0, stream>>>(v0, (long)M * 64);
  vfe_zero_f32<<<gr((long)M * 128), B, 0, stream>>>(outF, (long)M * 128);
  long padH = (long)(Np - N) * 128;
  if (padH > 0)
    vfe_zero_f16<<<gr(padH), B, 0, stream>>>(x1 + (size_t)N * 128, padH);

  // 2. repack weights into wave32 B-operand layout
  vfe_pack_w0<<<gr(2048), B, 0, stream>>>(W0, W0s);
  vfe_pack_w1<<<gr(16384), B, 0, stream>>>(W1, W1s);

  // 3. scatter-mean accumulation
  vfe_scatter_sum<<<gr(N), B, 0, stream>>>(feats, seg, vsum, vcnt, N);

  // 4. build 11-dim features, K-padded to 32, f16
  vfe_build_x<<<gr(Np), B, 0, stream>>>(feats, coors, seg, vsum, vcnt, x16, N, Np);

  // 5-7. layer 0: stats pass -> BN params -> apply pass (+scatter-max into v0)
  dim3 g1grid((rowTiles + 3) / 4, 4);
  vfe_gemm1_stats<<<g1grid, 128, 0, stream>>>(x16, W0s, stats0, stats0 + 64, rowTiles);
  vfe_bnparams<<<1, 64, 0, stream>>>(stats0, g0, b0, a0p, c0p, 64, (float)N);
  vfe_gemm1_apply<<<g1grid, 128, 0, stream>>>(x16, W0s, a0p, c0p, seg, x1, v0, N, rowTiles);

  // 8. concat gathered voxel max back into x1[:,64:128]
  vfe_gather_v0<<<gr((long)N * 64), B, 0, stream>>>(v0, seg, x1, N);

  // 9-11. layer 1: stats -> BN params -> apply (+scatter-max into output)
  dim3 g2grid((rowTiles + 3) / 4, 8);
  vfe_gemm2_stats<<<g2grid, 128, 0, stream>>>(x1, W1s, stats1, stats1 + 128, rowTiles);
  vfe_bnparams<<<1, 128, 0, stream>>>(stats1, g1, b1, a1p, c1p, 128, (float)N);
  vfe_gemm2_apply<<<g2grid, 128, 0, stream>>>(x1, W1s, a1p, c1p, seg, outF, N, rowTiles);

  // 12. voxel coords passthrough
  vfe_copy_coors<<<gr((long)M * 4), B, 0, stream>>>(vcrs, outC, M * 4);
}
